// AnaphoricityScorer_68264210202743
// MI455X (gfx1250) — compile-verified, matrix operand order checked by reading, not run on hardware
//
#include <hip/hip_runtime.h>
#include <hip/hip_bf16.h>

// ---------------------------------------------------------------------------
// Problem constants (match the reference)
// ---------------------------------------------------------------------------
#define NN 2048       // sequence length
#define EE 1024       // embedding dim
#define LL 4          // layers
#define E3 (3 * EE)   // 3072

typedef __attribute__((ext_vector_type(16))) __bf16 v16bf;
typedef __attribute__((ext_vector_type(8)))  __bf16 v8bf;
typedef __attribute__((ext_vector_type(8)))  float  v8f;

// round-to-nearest-even f32 -> bf16
static __device__ __forceinline__ __bf16 f32_to_bf16(float f) {
    unsigned u = __builtin_bit_cast(unsigned, f);
    u += 0x7FFFu + ((u >> 16) & 1u);
    unsigned short h = (unsigned short)(u >> 16);
    return __builtin_bit_cast(__bf16, h);
}

// ---------------------------------------------------------------------------
// f32 -> bf16 conversion (grid-stride)
// ---------------------------------------------------------------------------
__global__ void k_cvt_bf16(const float* __restrict__ in, __bf16* __restrict__ out, int n) {
    int i = blockIdx.x * blockDim.x + threadIdx.x;
    int stride = gridDim.x * blockDim.x;
    for (; i < n; i += stride) out[i] = f32_to_bf16(in[i]);
}

// ---------------------------------------------------------------------------
// 32x32 LDS-tiled bf16 transpose: dst[c][r] = src[r][c].
// Used once per layer on v so the w@v GEMM gets contiguous B fragments.
// block (32,8), grid (cols/32, rows/32).
// ---------------------------------------------------------------------------
__global__ __launch_bounds__(256)
void k_transpose_bf16(const __bf16* __restrict__ src, int srcLd,
                      __bf16* __restrict__ dst, int dstLd) {
    __shared__ __bf16 tile[32][33];   // +1 pad to dodge bank conflicts
    const int tx = threadIdx.x, ty = threadIdx.y;
    const int c0 = blockIdx.x * 32, r0 = blockIdx.y * 32;
#pragma unroll
    for (int i = 0; i < 32; i += 8)
        tile[ty + i][tx] = src[(size_t)(r0 + ty + i) * srcLd + c0 + tx];
    __syncthreads();
#pragma unroll
    for (int i = 0; i < 32; i += 8)
        dst[(size_t)(c0 + ty + i) * dstLd + r0 + tx] = tile[tx][ty + i];
}

// ---------------------------------------------------------------------------
// bf16 fragment loader for v_wmma_f32_16x16x32_bf16 (wave32).
// A-fragment (16x32, MxK): lane L -> row m = L&15, khalf = L>>4;
//   VGPR0-3 : K = khalf*8 .. +7     (16 contiguous bytes)
//   VGPR4-7 : K = 16+khalf*8 .. +7  (16 contiguous bytes)
// B-fragment (32x16, KxN) mirrors this with lane = column n, so for
// C = A @ B^T every fragment is two contiguous b128 loads from a row.
// ---------------------------------------------------------------------------
static __device__ __forceinline__ v16bf load_frag(const __bf16* __restrict__ p) {
    v8bf lo = *(const v8bf*)(p);
    v8bf hi = *(const v8bf*)(p + 16);
    v16bf r;
#pragma unroll
    for (int t = 0; t < 8; ++t) { r[t] = lo[t]; r[8 + t] = hi[t]; }
    return r;
}

// ---------------------------------------------------------------------------
// Tiled bf16 WMMA GEMM (f32 accumulate):  C[m,n] = sum_k A[m,k] * B[n,k]
// (both operands row-major; B holds the "transposed" operand's rows).
//   F32OUT=true : C = alpha*acc, causal mask -> f32   (scores path)
//   F32OUT=false: C = acc + bias            -> bf16   (activation path)
// Block = 256 threads = 8 waves, 4 (rows) x 2 (cols); each wave owns a
// 32x64 tile (2x4 WMMA accumulators) -> block tile 128x128.
// 8 WMMAs per 12 b128 loads per K-step. All dims divide exactly.
// ---------------------------------------------------------------------------
template <bool F32OUT>
__global__ __launch_bounds__(256)
void k_gemm_bf16_wmma(const __bf16* __restrict__ A, int lda,
                      const __bf16* __restrict__ B, int ldb,
                      float* __restrict__ Cf, __bf16* __restrict__ Cb, int ldc,
                      const float* __restrict__ bias, float alpha,
                      int K, int causalMask, int causalKLimit) {
    const int lane    = threadIdx.x & 31;
    const int wave    = threadIdx.x >> 5;
    const int rowBase = blockIdx.x * 128 + (wave >> 1) * 32;
    const int colBase = blockIdx.y * 128 + (wave & 1) * 64;
    const int ml      = lane & 15;   // row (A) / column (B) within 16-tile
    const int kh      = lane >> 4;   // K-half selector

    v8f acc[2][4] = {};

    int kEnd = K;
    if (causalKLimit) {              // w[m,k]==0 for k>m: skip dead K tiles
        int lim = blockIdx.x * 128 + 128;
        kEnd = lim < K ? lim : K;
    }

    const __bf16* Arow[2];
    Arow[0] = A + (size_t)(rowBase + ml) * lda;
    Arow[1] = A + (size_t)(rowBase + 16 + ml) * lda;
    const __bf16* Brow[4];
#pragma unroll
    for (int tj = 0; tj < 4; ++tj)
        Brow[tj] = B + (size_t)(colBase + tj * 16 + ml) * ldb;

    for (int k0 = 0; k0 < kEnd; k0 += 32) {
        const int ka = k0 + kh * 8;

        v16bf afrag[2];
#pragma unroll
        for (int ti = 0; ti < 2; ++ti) afrag[ti] = load_frag(Arow[ti] + ka);
        v16bf bfrag[4];
#pragma unroll
        for (int tj = 0; tj < 4; ++tj) bfrag[tj] = load_frag(Brow[tj] + ka);

        if (k0 + 32 < kEnd) {
            __builtin_prefetch(Arow[0] + ka + 32, 0, 1);   // global_prefetch_b8
            __builtin_prefetch(Brow[0] + ka + 32, 0, 1);
        }

#pragma unroll
        for (int ti = 0; ti < 2; ++ti)
#pragma unroll
            for (int tj = 0; tj < 4; ++tj)
                acc[ti][tj] = __builtin_amdgcn_wmma_f32_16x16x32_bf16(
                    false, afrag[ti], false, bfrag[tj],
                    (short)0, acc[ti][tj], false, false);
    }

    const float NEG_INF = -__builtin_inff();

    // C/D layout: VGPR r -> lanes 0-15: M=r, N=lane ; lanes 16-31: M=8+r, N=lane-16
#pragma unroll
    for (int tj = 0; tj < 4; ++tj) {
        const int col = colBase + tj * 16 + ml;
        float bv = 0.0f;
        if (!F32OUT && bias) bv = bias[col];   // one load per column tile
#pragma unroll
        for (int ti = 0; ti < 2; ++ti) {
            const int r0 = rowBase + ti * 16 + kh * 8;
#pragma unroll
            for (int r = 0; r < 8; ++r) {
                const int row = r0 + r;
                float val = acc[ti][tj][r];
                if (F32OUT) {
                    val *= alpha;
                    if (causalMask && col > row) val = NEG_INF;
                    Cf[(size_t)row * ldc + col] = val;
                } else {
                    Cb[(size_t)row * ldc + col] = f32_to_bf16(val + bv);
                }
            }
        }
    }
}

// ---------------------------------------------------------------------------
// Row softmax (masked entries arrive as -inf).
// One block (256 threads) per row of 2048 -> 8 elements/thread in registers.
// Writes bf16 w for the w@v GEMM and accumulates 0.25*w into d_out
// (layer 0 stores -> also initializes d_out; later layers add).
// ---------------------------------------------------------------------------
__global__ __launch_bounds__(256)
void k_softmax_row(const float* __restrict__ scores, __bf16* __restrict__ wb,
                   float* __restrict__ out, int firstLayer) {
    const int row = blockIdx.x;
    const int tid = threadIdx.x;
    const float* s = scores + (size_t)row * NN;

    float v[8];
    float m = -__builtin_inff();
#pragma unroll
    for (int i = 0; i < 8; ++i) {
        v[i] = s[tid + i * 256];
        m = fmaxf(m, v[i]);
    }

    __shared__ float red[256];
    red[tid] = m;
    __syncthreads();
    for (int off = 128; off > 0; off >>= 1) {
        if (tid < off) red[tid] = fmaxf(red[tid], red[tid + off]);
        __syncthreads();
    }
    const float rowmax = red[0];
    __syncthreads();

    float sum = 0.0f;
#pragma unroll
    for (int i = 0; i < 8; ++i) {
        v[i] = __expf(v[i] - rowmax);   // exp(-inf) = 0 handles the mask
        sum += v[i];
    }
    red[tid] = sum;
    __syncthreads();
    for (int off = 128; off > 0; off >>= 1) {
        if (tid < off) red[tid] += red[tid + off];
        __syncthreads();
    }
    const float inv = 1.0f / red[0];

#pragma unroll
    for (int i = 0; i < 8; ++i) {
        const int n = tid + i * 256;
        const float w = v[i] * inv;
        const size_t idx = (size_t)row * NN + n;
        wb[idx] = f32_to_bf16(w);
        const float contrib = 0.25f * w;   // mean over L=4 layers
        if (firstLayer) out[idx] = contrib;
        else            out[idx] += contrib;
    }
}

// ---------------------------------------------------------------------------
// Host-side orchestration
// ---------------------------------------------------------------------------
extern "C" void kernel_launch(void* const* d_in, const int* in_sizes, int n_in,
                              void* d_out, int out_size, void* d_ws, size_t ws_size,
                              hipStream_t stream) {
    (void)in_sizes; (void)n_in; (void)out_size; (void)ws_size;

    const float* x_f    = (const float*)d_in[0];   // [N, E]
    const float* Wqkv_f = (const float*)d_in[1];   // [L, 3E, E]
    const float* bqkv_f = (const float*)d_in[2];   // [L, 3E]
    const float* Wo_f   = (const float*)d_in[3];   // [L, E, E]
    const float* bo_f   = (const float*)d_in[4];   // [L, E]
    float* out          = (float*)d_out;           // [N, N]

    // ---- workspace layout (~80 MB), 256B-aligned segments ----
    char* base = (char*)d_ws;
    size_t off = 0;
    auto take = [&](size_t bytes) {
        void* p = base + off;
        off += (bytes + 255) & ~(size_t)255;
        return p;
    };
    __bf16* xb     = (__bf16*)take((size_t)NN * EE * 2);        // current x (bf16)
    __bf16* qkvb   = (__bf16*)take((size_t)NN * E3 * 2);        // qkv (bf16)
    __bf16* vT     = (__bf16*)take((size_t)EE * NN * 2);        // v transposed (bf16)
    float*  scores = (float*) take((size_t)NN * NN * 4);        // scores (f32)
    __bf16* wbb    = (__bf16*)take((size_t)NN * NN * 2);        // softmax(w) (bf16)
    __bf16* ctxb   = (__bf16*)take((size_t)NN * EE * 2);        // w@v (bf16)
    __bf16* wqb    = (__bf16*)take((size_t)LL * E3 * EE * 2);   // Wqkv (bf16)
    __bf16* wob    = (__bf16*)take((size_t)LL * EE * EE * 2);   // Wo   (bf16)

    const dim3 blk(256);

    // ---- one-time f32 -> bf16 conversions ----
    k_cvt_bf16<<<2048, blk, 0, stream>>>(x_f,    xb,  NN * EE);
    k_cvt_bf16<<<4096, blk, 0, stream>>>(Wqkv_f, wqb, LL * E3 * EE);
    k_cvt_bf16<<<4096, blk, 0, stream>>>(Wo_f,   wob, LL * EE * EE);

    const float scale = 0.03125f;   // 1/sqrt(E) = 1/32

    for (int i = 0; i < LL; ++i) {
        const __bf16* Wq = wqb + (size_t)i * E3 * EE;
        const __bf16* Wo = wob + (size_t)i * EE * EE;
        const float*  bq = bqkv_f + (size_t)i * E3;
        const float*  bo = bo_f   + (size_t)i * EE;

        // 1) qkv[N,3E] = x @ Wqkv^T + bqkv            (bf16 out)
        {
            dim3 grid(NN / 128, E3 / 128);
            k_gemm_bf16_wmma<false><<<grid, blk, 0, stream>>>(
                xb, EE, Wq, EE, nullptr, qkvb, E3, bq, 1.0f, EE, 0, 0);
        }
        // 1b) vT[E,N] = v^T  (one LDS transpose so w@v gets contiguous B rows)
        {
            dim3 tgrid(EE / 32, NN / 32), tblk(32, 8);
            k_transpose_bf16<<<tgrid, tblk, 0, stream>>>(qkvb + 2 * EE, E3, vT, NN);
        }
        // 2) scores[N,N] = scale * (q @ k^T) + causal mask   (f32 out)
        {
            dim3 grid(NN / 128, NN / 128);
            k_gemm_bf16_wmma<true><<<grid, blk, 0, stream>>>(
                qkvb /*q*/, E3, qkvb + EE /*k rows*/, E3,
                scores, nullptr, NN, nullptr, scale, EE, 1, 0);
        }
        // 3) w = softmax(scores); d_out (+)= 0.25*w; w -> bf16
        k_softmax_row<<<NN, blk, 0, stream>>>(scores, wbb, out, i == 0);

        // 4) ctx[N,E] = w @ v = w @ (vT)^T   (bf16 out; causal K-limit)
        {
            dim3 grid(NN / 128, EE / 128);
            k_gemm_bf16_wmma<false><<<grid, blk, 0, stream>>>(
                wbb, NN, vT, NN, nullptr, ctxb, EE, nullptr, 1.0f, NN, 0, 1);
        }
        // 5) x = ctx @ Wo^T + bo   (bf16 out, next layer's input)
        {
            dim3 grid(NN / 128, EE / 128);
            k_gemm_bf16_wmma<false><<<grid, blk, 0, stream>>>(
                ctxb, EE, Wo, EE, nullptr, xb, EE, bo, 1.0f, EE, 0, 0);
        }
    }
}
